// NoiseCancellingAttentionMechanism_24670292148716
// MI455X (gfx1250) — matrix operand order repaired and checked by reading, not compiled, not run on hardware
//
#include <hip/hip_runtime.h>
#include <hip/hip_bf16.h>

// ---------------------------------------------------------------------------
// Differential (noise-cancelling) attention for MI455X / gfx1250, wave32.
// bf16 WMMA (v_wmma_f32_16x16x32_bf16) everywhere; flash-style dual online
// softmax (scores never materialized); K/V tiles staged with CDNA5 async
// global->LDS DMA (global_load_async_to_lds_b64 + s_wait_asynccnt), double
// buffered so the copy for tile i+1 overlaps tile i's WMMAs.
// ---------------------------------------------------------------------------

typedef __bf16 bf16_t;
typedef __attribute__((ext_vector_type(16))) __bf16 v16bf;
typedef __attribute__((ext_vector_type(8)))  float  v8f;

#define DEV __device__ __forceinline__

static constexpr int   Bb   = 2, Ss = 2048, Dd = 1024, Hh = 16, DHd = 64;
static constexpr int   NTOK = Bb * Ss;   // 4096
static constexpr float SCALE = 0.125f;   // 1/sqrt(64)

DEV unsigned short f2bf_bits(float f) {
  unsigned u = __float_as_uint(f);
  unsigned r = u + 0x7FFFu + ((u >> 16) & 1u);  // round-to-nearest-even
  return (unsigned short)(r >> 16);
}
DEV bf16_t to_bf16(float f) {
  unsigned short h = f2bf_bits(f);
  return __builtin_bit_cast(bf16_t, h);
}
DEV unsigned pack2(float a, float b) {
  return (unsigned)f2bf_bits(a) | ((unsigned)f2bf_bits(b) << 16);
}
DEV v8f zero8() { v8f z = {0.f, 0.f, 0.f, 0.f, 0.f, 0.f, 0.f, 0.f}; return z; }

// ---- CDNA5 async global->LDS copy (ISA 15.18.3 op 97, ASYNCcnt tracked) ----
DEV void async_b64(unsigned lds_dst, const void* gsrc) {
  asm volatile("global_load_async_to_lds_b64 %0, %1, off"
               :: "v"(lds_dst), "v"(gsrc) : "memory");
}
DEV void wait_async0() { asm volatile("s_wait_asynccnt 0x0" ::: "memory"); }
DEV unsigned lds_addr(const void* p) { return (unsigned)(size_t)p; }

// A fragment (16x32 bf16): rows on lanes 0..15; lane-group g selects K-halves.
// ISA 7.12.2: e in [0,8): K = g*8+e ; e in [8,16): K = 16 + g*8 + (e-8).
DEV v16bf load_a_frag(const bf16_t* p, int ldk) {
  const int lane = threadIdx.x & 31;
  const int row = lane & 15, g = lane >> 4;
  const bf16_t* q = p + row * ldk + g * 8;
  v16bf a;
#pragma unroll
  for (int e = 0; e < 8; ++e) a[e] = q[e];
#pragma unroll
  for (int e = 0; e < 8; ++e) a[8 + e] = q[16 + e];
  return a;
}

// B fragment (32x16 bf16): cols on lanes 0..15; e<8: K = g*8+e; e>=8: K = 16+g*8+(e-8).
DEV v16bf load_b_frag(const bf16_t* p, int stride_k, int stride_n) {
  const int lane = threadIdx.x & 31;
  const int n = lane & 15, g = lane >> 4;
  const bf16_t* q = p + n * stride_n + g * 8 * stride_k;
  v16bf b;
#pragma unroll
  for (int e = 0; e < 8; ++e) b[e] = q[e * stride_k];
#pragma unroll
  for (int e = 0; e < 8; ++e) b[8 + e] = q[(16 + e) * stride_k];
  return b;
}

DEV v8f wmma_bf16(v16bf a, v16bf b, v8f c) {
  return __builtin_amdgcn_wmma_f32_16x16x32_bf16(false, a, false, b,
                                                 (short)0, c, false, false);
}

// ---------------------------------------------------------------------------
// GEMM  C[M,N] = A[M,K] @ W[K,N] + bias, bf16 WMMA, f32 accumulate.
// A_BF16: A is bf16 (already converted); else fp32 converted while staging.
// HEAD_SCATTER: write bf16 into [B,H,S,DH] layout; else write f32 row-major.
// ---------------------------------------------------------------------------
template <bool A_BF16, bool HEAD_SCATTER>
__global__ __launch_bounds__(256) void gemm_bias_kernel(
    const void* __restrict__ Ap, const float* __restrict__ W,
    const float* __restrict__ bias, void* __restrict__ Cp,
    int M, int Kdim, int N) {
  constexpr int BM = 128, BN = 128, BK = 64;
  constexpr int LDA = BK + 8;   // pad: bank-conflict-free strided reads
  constexpr int LDB = BN + 8;
  __shared__ bf16_t As[BM][LDA];
  __shared__ bf16_t Bs[BK][LDB];

  const int tid = threadIdx.x;
  const int wave = tid >> 5;
  const int m0 = blockIdx.x * BM, n0 = blockIdx.y * BN;

  v8f acc[8];
#pragma unroll
  for (int c = 0; c < 8; ++c) acc[c] = zero8();

  for (int k0 = 0; k0 < Kdim; k0 += BK) {
    __syncthreads();
    // ---- stage A tile (128x64) as bf16 ----
    if (A_BF16) {
      const bf16_t* A = (const bf16_t*)Ap;
#pragma unroll
      for (int q = 0; q < 8; ++q) {
        const int fi = tid * 8 + q;              // 2048 uint2 chunks (4 bf16)
        const int row = fi >> 4, cq = (fi & 15) * 4;
        *(uint2*)&As[row][cq] =
            *(const uint2*)(A + (size_t)(m0 + row) * Kdim + k0 + cq);
      }
    } else {
      const float* A = (const float*)Ap;
#pragma unroll
      for (int q = 0; q < 8; ++q) {
        const int fi = tid * 8 + q;              // 2048 float4 chunks
        const int row = fi >> 4, cq = (fi & 15) * 4;
        const float4 v = *(const float4*)(A + (size_t)(m0 + row) * Kdim + k0 + cq);
        uint2 pk; pk.x = pack2(v.x, v.y); pk.y = pack2(v.z, v.w);
        *(uint2*)&As[row][cq] = pk;
      }
    }
    // ---- stage W tile (64x128) as bf16 ----
#pragma unroll
    for (int q = 0; q < 8; ++q) {
      const int fi = tid * 8 + q;                // 2048 float4 chunks
      const int row = fi >> 5, nq = (fi & 31) * 4;
      const float4 v = *(const float4*)(W + (size_t)(k0 + row) * N + n0 + nq);
      uint2 pk; pk.x = pack2(v.x, v.y); pk.y = pack2(v.z, v.w);
      *(uint2*)&Bs[row][nq] = pk;
    }
    if (k0 + BK < Kdim)  // hint next W tile into cache (global_prefetch)
      __builtin_prefetch(W + (size_t)(k0 + BK + (tid >> 2)) * N + n0, 0, 1);
    __syncthreads();

#pragma unroll
    for (int kk = 0; kk < BK; kk += 32) {
      const v16bf a = load_a_frag(&As[wave * 16][kk], LDA);
#pragma unroll
      for (int c = 0; c < 8; ++c) {
        const v16bf b = load_b_frag(&Bs[kk][c * 16], LDB, 1);
        acc[c] = wmma_bf16(a, b, acc[c]);
      }
    }
  }

  const int lane = tid & 31;
  const int n_in = lane & 15, g = lane >> 4;
#pragma unroll
  for (int c = 0; c < 8; ++c) {
    const int n = n0 + c * 16 + n_in;
    const float bv = bias[n];
#pragma unroll
    for (int r = 0; r < 8; ++r) {
      const int m = m0 + wave * 16 + r + 8 * g;
      const float val = acc[c][r] + bv;
      if (HEAD_SCATTER) {
        // token m -> (b,s); column n -> (h,d); layout [B,H,S,DH], bf16
        bf16_t* C = (bf16_t*)Cp;
        const int b = m >> 11, s = m & 2047;
        const int h = n >> 6, d = n & 63;
        C[((size_t)((b * Hh + h) * Ss + s)) * DHd + d] = to_bf16(val);
      } else {
        ((float*)Cp)[(size_t)m * N + n] = val;
      }
    }
  }
}

// ---------------------------------------------------------------------------
// Dual online-softmax update for one 16x64 score tile (4 n-chunks of v8f).
// C/D layout: lane holds column n=lane%16; reg r holds row r+8*(lane/16).
// Row reductions are shuffles across the 16-lane half-group.
// ---------------------------------------------------------------------------
DEV void online_update(v8f (&s)[4], float (&m)[8], float (&l)[8], v8f (&acc)[4]) {
#pragma unroll
  for (int r = 0; r < 8; ++r) {
    float mx = fmaxf(fmaxf(s[0][r], s[1][r]), fmaxf(s[2][r], s[3][r]));
#pragma unroll
    for (int off = 1; off < 16; off <<= 1) mx = fmaxf(mx, __shfl_xor(mx, off, 32));
    mx *= SCALE;
    const float nm = fmaxf(m[r], mx);
    const float corr = __expf(m[r] - nm);
    m[r] = nm;
    float rs = 0.f;
#pragma unroll
    for (int c = 0; c < 4; ++c) {
      const float p = __expf(s[c][r] * SCALE - nm);
      s[c][r] = p;
      rs += p;
    }
#pragma unroll
    for (int off = 1; off < 16; off <<= 1) rs += __shfl_xor(rs, off, 32);
    l[r] = l[r] * corr + rs;
#pragma unroll
    for (int c = 0; c < 4; ++c) acc[c][r] *= corr;
  }
}

// ---------------------------------------------------------------------------
// Flash-style differential attention. Block = 128 query rows of one (b,h);
// 8 waves x 16 rows. K/V 64x64 tiles double-buffered via async DMA and shared
// by both score streams. P tiles are strictly per-wave (each wave writes and
// reads only its own 16 rows), so P needs no barriers and one buffer serves
// P1 then P2.   Out = P1@V / l1 - lam * P2@V / l2.
// ---------------------------------------------------------------------------
__global__ __launch_bounds__(256) void diff_attn_kernel(
    const bf16_t* __restrict__ Q1, const bf16_t* __restrict__ Q2,
    const bf16_t* __restrict__ Kh, const bf16_t* __restrict__ Vh,
    const float* __restrict__ lam_p, bf16_t* __restrict__ Out) {
  constexpr int QT = 128, KT = 64;
  constexpr int LDK = DHd + 8;  // 72: conflict-free strided frag reads, 8B rows
  constexpr int LDP = DHd + 4;  // 68: 34-dword row stride, bijective mod 64
  __shared__ bf16_t Ks[2][KT][LDK];
  __shared__ bf16_t Vs[2][KT][LDK];
  __shared__ bf16_t Ps[QT][LDP];

  const int tid = threadIdx.x, wave = tid >> 5, lane = tid & 31;
  const int g = lane >> 4, n_in = lane & 15;
  const int bh = blockIdx.y;                // 0..31
  const int b = bh >> 4, h = bh & 15;
  const int q0 = blockIdx.x * QT;
  const size_t head = (size_t)bh * Ss * DHd;
  const bf16_t* kbase = Kh + head;
  const bf16_t* vbase = Vh + head;

  // each thread owns 4 8-byte chunks of each 64x64 tile
  const int st_row = (tid * 4) >> 4;          // rows 0..63 (all 4 chunks share one row)
  // (recompute per-chunk inside the stager instead)

  // Q fragments for this wave's 16 rows live in registers for the whole sweep.
  const bf16_t* q1p = Q1 + head + (size_t)(q0 + wave * 16) * DHd;
  const bf16_t* q2p = Q2 + head + (size_t)(q0 + wave * 16) * DHd;
  v16bf q1f[2], q2f[2];
  q1f[0] = load_a_frag(q1p, DHd);      q1f[1] = load_a_frag(q1p + 32, DHd);
  q2f[0] = load_a_frag(q2p, DHd);      q2f[1] = load_a_frag(q2p + 32, DHd);

  v8f O1[4], O2[4];
  float m1[8], l1[8], m2[8], l2[8];
#pragma unroll
  for (int c = 0; c < 4; ++c) { O1[c] = zero8(); O2[c] = zero8(); }
#pragma unroll
  for (int r = 0; r < 8; ++r) { m1[r] = m2[r] = -1e30f; l1[r] = l2[r] = 0.f; }

  // async-stage one K/V tile pair into buffer `buf`
  auto stage = [&](int buf, int kt) {
#pragma unroll
    for (int q = 0; q < 4; ++q) {
      const int fi = tid * 4 + q;              // 1024 8B chunks per tile
      const int row = fi >> 4, cq = (fi & 15) * 4;
      async_b64(lds_addr(&Ks[buf][row][cq]),
                kbase + (size_t)(kt + row) * DHd + cq);
      async_b64(lds_addr(&Vs[buf][row][cq]),
                vbase + (size_t)(kt + row) * DHd + cq);
    }
  };

  stage(0, 0);                                  // prologue

  for (int kt = 0; kt < Ss; kt += KT) {
    const int cur = (kt >> 6) & 1;
    wait_async0();        // my chunks landed in LDS
    __syncthreads();      // everyone's chunks landed; prev-iter reads done
    if (kt + KT < Ss) stage(cur ^ 1, kt + KT);  // DMA next under this compute

    // s1 = Q1 Kt^T, s2 = Q2 Kt^T   (B frag: lane = key row, e = d contiguous)
    v8f s1[4], s2[4];
#pragma unroll
    for (int c = 0; c < 4; ++c) { s1[c] = zero8(); s2[c] = zero8(); }
#pragma unroll
    for (int kk = 0; kk < 64; kk += 32) {
      const int ki = kk >> 5;
#pragma unroll
      for (int c = 0; c < 4; ++c) {
        const v16bf bf = load_b_frag(&Ks[cur][c * 16][kk], 1, LDK);
        s1[c] = wmma_bf16(q1f[ki], bf, s1[c]);
        s2[c] = wmma_bf16(q2f[ki], bf, s2[c]);
      }
    }

    online_update(s1, m1, l1, O1);
    online_update(s2, m2, l2, O2);

    // P1 -> O1 (per-wave rows of Ps; no cross-wave sharing -> no barrier)
#pragma unroll
    for (int c = 0; c < 4; ++c)
#pragma unroll
      for (int r = 0; r < 8; ++r)
        Ps[wave * 16 + r + 8 * g][c * 16 + n_in] = to_bf16(s1[c][r]);
#pragma unroll
    for (int kk = 0; kk < 64; kk += 32) {
      const v16bf a1 = load_a_frag(&Ps[wave * 16][kk], LDP);
#pragma unroll
      for (int c = 0; c < 4; ++c) {
        const v16bf bv = load_b_frag(&Vs[cur][kk][c * 16], LDK, 1);
        O1[c] = wmma_bf16(a1, bv, O1[c]);
      }
    }

    // P2 -> O2 (reuse Ps; in-wave LDS ordering guarantees RAW/WAR safety)
#pragma unroll
    for (int c = 0; c < 4; ++c)
#pragma unroll
      for (int r = 0; r < 8; ++r)
        Ps[wave * 16 + r + 8 * g][c * 16 + n_in] = to_bf16(s2[c][r]);
#pragma unroll
    for (int kk = 0; kk < 64; kk += 32) {
      const v16bf a2 = load_a_frag(&Ps[wave * 16][kk], LDP);
#pragma unroll
      for (int c = 0; c < 4; ++c) {
        const v16bf bv = load_b_frag(&Vs[cur][kk][c * 16], LDK, 1);
        O2[c] = wmma_bf16(a2, bv, O2[c]);
      }
    }
  }

  const float lam = *lam_p;
#pragma unroll
  for (int r = 0; r < 8; ++r) {
    const float i1 = 1.f / l1[r];
    const float i2 = lam / l2[r];
    const int token = b * Ss + q0 + wave * 16 + r + 8 * g;
#pragma unroll
    for (int c = 0; c < 4; ++c) {
      const int col = h * DHd + c * 16 + n_in;
      Out[(size_t)token * Dd + col] = to_bf16(O1[c][r] * i1 - O2[c][r] * i2);
    }
  }
}

// ---------------------------------------------------------------------------
extern "C" void kernel_launch(void* const* d_in, const int* in_sizes, int n_in,
                              void* d_out, int out_size, void* d_ws,
                              size_t ws_size, hipStream_t stream) {
  (void)in_sizes; (void)n_in; (void)out_size; (void)ws_size;
  const float* v   = (const float*)d_in[0];
  const float* k   = (const float*)d_in[1];
  const float* q1  = (const float*)d_in[2];
  const float* q2  = (const float*)d_in[3];
  const float* Wq  = (const float*)d_in[4];
  const float* bq  = (const float*)d_in[5];
  const float* Wk  = (const float*)d_in[6];
  const float* bk  = (const float*)d_in[7];
  const float* Wv  = (const float*)d_in[8];
  const float* bv  = (const float*)d_in[9];
  const float* Wo  = (const float*)d_in[10];
  const float* bo  = (const float*)d_in[11];
  const float* lam = (const float*)d_in[12];
  float* out = (float*)d_out;

  const size_t plane = (size_t)NTOK * Dd * sizeof(bf16_t);  // 8 MB each
  char* ws = (char*)d_ws;
  bf16_t* q1h = (bf16_t*)(ws + 0 * plane);  // [B,H,S,DH] bf16
  bf16_t* q2h = (bf16_t*)(ws + 1 * plane);
  bf16_t* kh  = (bf16_t*)(ws + 2 * plane);
  bf16_t* vh  = (bf16_t*)(ws + 3 * plane);
  bf16_t* att = (bf16_t*)(ws + 4 * plane);  // [NTOK, D] bf16

  const dim3 gg(NTOK / 128, Dd / 128);      // 32 x 8 blocks
  gemm_bias_kernel<false, true><<<gg, 256, 0, stream>>>(q1, Wq, bq, q1h, NTOK, Dd, Dd);
  gemm_bias_kernel<false, true><<<gg, 256, 0, stream>>>(q2, Wq, bq, q2h, NTOK, Dd, Dd);
  gemm_bias_kernel<false, true><<<gg, 256, 0, stream>>>(k,  Wk, bk, kh,  NTOK, Dd, Dd);
  gemm_bias_kernel<false, true><<<gg, 256, 0, stream>>>(v,  Wv, bv, vh,  NTOK, Dd, Dd);

  diff_attn_kernel<<<dim3(Ss / 128, Bb * Hh), 256, 0, stream>>>(q1h, q2h, kh, vh, lam, att);

  gemm_bias_kernel<true, false><<<gg, 256, 0, stream>>>(att, Wo, bo, out, NTOK, Dd, Dd);
}
